// CharLSTM_4776003633159
// MI455X (gfx1250) — matrix operand order repaired and checked by reading, not compiled, not run on hardware
//
#include <hip/hip_runtime.h>

// ---------------------------------------------------------------------------
// Char-LSTM for MI455X (gfx1250, wave32, WMMA + TDM).
//   EMB=512, HID=2048, T=4096, D=2560, GATE=4*HID=8192.
// Pipeline:
//   1) convert weights -> bf16, transposed, gate-interleaved col = 4*j + g
//   2) embed X[T,512] (bf16)
//   3) Xp[T,8192] = X @ Wx + bias         (WMMA bf16 GEMM, batched over time)
//   4) persistent recurrent kernel: h staged into LDS via TENSOR_LOAD_TO_LDS
//      (TDM), h @ Wh matvec via WMMA, LSTM cell update, grid barrier per step
//   5) out[T,256] = hs @ Wout + bout      (WMMA bf16 GEMM)
// ---------------------------------------------------------------------------

#define EMBW   512
#define HIDW   2048
#define TLEN   4096
#define DIN    2560          // EMB + HID
#define GATE   8192          // 4 * HID
#define NCH    256
#define NBLK   64            // persistent recurrent blocks

typedef __attribute__((ext_vector_type(16))) __bf16 v16bf;
typedef __attribute__((ext_vector_type(8)))  __bf16 v8bf;
typedef __attribute__((ext_vector_type(8)))  float  v8f;
typedef __attribute__((ext_vector_type(4)))  unsigned int v4u;
typedef __attribute__((ext_vector_type(8)))  int v8i;
typedef __attribute__((ext_vector_type(4)))  int v4i;

union Frag { v16bf v; v8bf h[2]; };

__device__ __forceinline__ v8f wmma_bf16(v8f acc, Frag a, Frag b) {
  // D = A(16x32 bf16) * B(32x16 bf16) + C(16x16 f32)
  return __builtin_amdgcn_wmma_f32_16x16x32_bf16(
      /*neg_a=*/false, a.v, /*neg_b=*/false, b.v,
      /*c_mod=*/(short)0, acc, /*reuse_a=*/false, /*reuse_b=*/false);
}

__device__ __forceinline__ float sigmoidf_(float x) {
  return 1.0f / (1.0f + __expf(-x));
}

// --------------------------- grid-wide barrier -----------------------------
__device__ __forceinline__ void grid_sync(unsigned* bar, unsigned nblocks) {
  __syncthreads();
  if (threadIdx.x == 0) {
    __threadfence();
    unsigned gen = __atomic_load_n(bar + 1, __ATOMIC_RELAXED);
    unsigned v   = __atomic_fetch_add(bar, 1u, __ATOMIC_ACQ_REL);
    if (v == nblocks - 1) {
      __atomic_store_n(bar, 0u, __ATOMIC_RELAXED);
      __atomic_fetch_add(bar + 1, 1u, __ATOMIC_RELEASE);
    } else {
      while (__atomic_load_n(bar + 1, __ATOMIC_ACQUIRE) == gen)
        __builtin_amdgcn_s_sleep(2);
    }
    __threadfence();
  }
  __syncthreads();
}

// ------------------------- weight conversion -------------------------------
// WxT[c][k] (c in [0,8192), k in [0,512)) and WhT[c][k] (k in [0,2048)),
// where c = 4*j + gate, gate order {f,i,o,g}. Source W is [2560,2048] row-major.
__global__ void k_convert_gates(const float* __restrict__ Wf, const float* __restrict__ Wi,
                                const float* __restrict__ Wo, const float* __restrict__ Wc,
                                __bf16* __restrict__ WxT, __bf16* __restrict__ WhT) {
  size_t idx = (size_t)blockIdx.x * blockDim.x + threadIdx.x;  // [0, GATE*DIN)
  int c = (int)(idx % GATE);
  int k = (int)(idx / GATE);
  int j = c >> 2, g = c & 3;
  const float* W = (g == 0) ? Wf : (g == 1) ? Wi : (g == 2) ? Wo : Wc;
  __bf16 v = (__bf16)W[(size_t)k * HIDW + j];
  if (k < EMBW) WxT[(size_t)c * EMBW + k]          = v;
  else          WhT[(size_t)c * HIDW + (k - EMBW)] = v;
}

__global__ void k_convert_wout(const float* __restrict__ Wout, __bf16* __restrict__ WoutT) {
  size_t idx = (size_t)blockIdx.x * blockDim.x + threadIdx.x;  // [0, NCH*HID)
  int n = (int)(idx % NCH);
  int k = (int)(idx / NCH);
  WoutT[(size_t)n * HIDW + k] = (__bf16)Wout[(size_t)k * NCH + n];
}

__global__ void k_embed(const int* __restrict__ seq, const float* __restrict__ emb,
                        __bf16* __restrict__ X) {
  size_t idx = (size_t)blockIdx.x * blockDim.x + threadIdx.x;  // [0, T*EMB)
  int t = (int)(idx / EMBW);
  int e = (int)(idx % EMBW);
  X[idx] = (__bf16)emb[(size_t)seq[t] * EMBW + e];
}

__global__ void k_init(const float* __restrict__ hidden, const float* __restrict__ cell,
                       __bf16* __restrict__ hbuf, float* __restrict__ cbuf,
                       unsigned* __restrict__ bar) {
  int i = blockIdx.x * blockDim.x + threadIdx.x;
  if (i < HIDW) {
    hbuf[i]        = (__bf16)hidden[i];
    hbuf[HIDW + i] = (__bf16)0.0f;
    cbuf[i]        = cell[i];
  }
  if (i == 0) { bar[0] = 0u; bar[1] = 0u; }
}

// --------------------- Xp = X @ WxT^T + bias (WMMA) ------------------------
// Block: 256 threads = 8 waves; wave -> one 16x16 output tile, K = 512.
__global__ void k_xproj(const __bf16* __restrict__ X, const __bf16* __restrict__ WxT,
                        const float* __restrict__ bfv, const float* __restrict__ biv,
                        const float* __restrict__ bov, const float* __restrict__ bcv,
                        __bf16* __restrict__ Xp) {
  int wave = threadIdx.x >> 5, lane = threadIdx.x & 31;
  int half = lane >> 4, l15 = lane & 15;
  int cbase = (blockIdx.x * 8 + wave) * 16;   // [0, 8192)
  int tbase = blockIdx.y * 16;                // [0, 4096)

  v8f acc = {};
  const __bf16* arow = X   + (size_t)(tbase + l15) * EMBW + half * 8;   // A: rows = t
  const __bf16* brow = WxT + (size_t)(cbase + l15) * EMBW + half * 8;   // B: cols = c
#pragma unroll 4
  for (int kc = 0; kc < EMBW / 32; ++kc) {
    Frag a, b;
    a.h[0] = *(const v8bf*)(arow);
    a.h[1] = *(const v8bf*)(arow + 16);
    b.h[0] = *(const v8bf*)(brow);
    b.h[1] = *(const v8bf*)(brow + 16);
    acc = wmma_bf16(acc, a, b);
    arow += 32; brow += 32;
  }
  int c = cbase + l15;
  int g = c & 3, j = c >> 2;
  const float* bsel = (g == 0) ? bfv : (g == 1) ? biv : (g == 2) ? bov : bcv;
  float bias = bsel[j];
#pragma unroll
  for (int q = 0; q < 8; ++q) {
    int row = tbase + q + half * 8;
    Xp[(size_t)row * GATE + c] = (__bf16)(acc[q] + bias);
  }
}

// --------------------- persistent recurrent kernel -------------------------
// 64 blocks x 256 threads. Block wg owns hidden units [32*wg, 32*wg+32)
// == gate columns [128*wg, 128*wg+128). Wave w owns 16 columns (4 units).
// h_{t-1} is staged into LDS by the Tensor Data Mover: wave 0 issues one
// TENSOR_LOAD_TO_LDS (1-D descriptor, 2048 x 2B, tile == tensor), waits on
// TENSORcnt, and __syncthreads publishes the LDS to the other 7 waves.
// Matvec trick: A = WhT tile (16 cols x 32 k), B = h broadcast across the 16
// N-lanes -> every D column identical; lanes 0 and 16 hold complete sums for
// cols [cbase..cbase+7] / [cbase+8..cbase+15] = 2 hidden units each (f,i,o,g
// adjacent thanks to the 4*j+g interleave) and do the cell update in-register.
__global__ void k_recurrent(const __bf16* __restrict__ WhT, const __bf16* __restrict__ Xp,
                            __bf16* __restrict__ hbuf,   // [2][HID] bf16, double buffer
                            float*  __restrict__ cbuf,   // [HID]
                            __bf16* __restrict__ hs,     // [T][HID] bf16
                            float*  __restrict__ hc_fin, // h_fin[HID], c_fin[HID]
                            unsigned* __restrict__ bar) {
  __shared__ __align__(16) __bf16 hsh[HIDW];
  const int tid  = threadIdx.x;
  const int wave = tid >> 5, lane = tid & 31;
  const int half = lane >> 4, l15 = lane & 15;
  const int cbase = blockIdx.x * 128 + wave * 16;
  const __bf16* wbase = WhT + (size_t)(cbase + l15) * HIDW + half * 8;

  // TDM descriptor pieces that are loop-invariant
  const unsigned lds_off = (unsigned)(size_t)(&hsh[0]);   // LDS byte offset
  v8i g1;
  g1[0] = 0x10000;                 // workgroup_mask=0, data_size=1 (2 bytes)
  g1[1] = (int)(HIDW << 16);       // tensor_dim0[15:0] at bits [63:48]
  g1[2] = 1 << 16;                 // tensor_dim0 hi = 0, tensor_dim1 = 1
  g1[3] = (int)(HIDW << 16);       // tile_dim0 = 2048 at bits [127:112]
  g1[4] = 1;                       // tile_dim1 = 1, tile_dim2 = 0
  g1[5] = HIDW;                    // tensor_dim0_stride = 2048
  g1[6] = 0;
  g1[7] = 0;
  const v4i gz4 = {0, 0, 0, 0};    // groups 2/3 unused (<= 2-D tensor)
  const v8i gz8 = {0, 0, 0, 0, 0, 0, 0, 0};

  for (int t = 0; t < TLEN; ++t) {
    // stage h_{t-1} into LDS (4 KB) with one TDM transfer from wave 0
    const __bf16* hsrc = hbuf + ((t & 1) ? HIDW : 0);
    if (wave == 0) {
      unsigned long long ga = (unsigned long long)(size_t)hsrc;
      v4u g0;
      g0.x = 1u;                                           // count=1, user mode
      g0.y = lds_off;                                      // lds_addr
      g0.z = (unsigned)ga;                                 // global_addr lo
      g0.w = ((unsigned)(ga >> 32) & 0x01FFFFFFu) | (2u << 30);  // addr hi | type=2
      __builtin_amdgcn_tensor_load_to_lds(g0, g1, gz4, gz4, gz8, 0);
      __builtin_amdgcn_s_wait_tensorcnt(0);
    }
    __syncthreads();

    v8f acc = {};
    const __bf16* wrow = wbase;
    const __bf16* hp   = hsh + half * 8;
#pragma unroll 8
    for (int kc = 0; kc < HIDW / 32; ++kc) {
      Frag a, b;
      a.h[0] = *(const v8bf*)(wrow);        // K {0..7} / {8..15}
      a.h[1] = *(const v8bf*)(wrow + 16);   // K {16..23} / {24..31}
      b.h[0] = *(const v8bf*)(hp);          // h broadcast across N lanes
      b.h[1] = *(const v8bf*)(hp + 16);
      acc = wmma_bf16(acc, a, b);
      wrow += 32; hp += 32;
    }

    if (l15 == 0) {  // lanes 0 and 16: complete column sums in acc[0..7]
      const __bf16* xp = Xp + (size_t)t * GATE + cbase + half * 8;
      if (t + 1 < TLEN)
        __builtin_prefetch(Xp + (size_t)(t + 1) * GATE + cbase + half * 8, 0, 1);
      float pre[8];
#pragma unroll
      for (int q = 0; q < 8; ++q) pre[q] = acc[q] + (float)xp[q];
      int j0 = (cbase >> 2) + half * 2;
#pragma unroll
      for (int u = 0; u < 2; ++u) {
        float fg = sigmoidf_(pre[u * 4 + 0]);
        float ig = sigmoidf_(pre[u * 4 + 1]);
        float og = sigmoidf_(pre[u * 4 + 2]);
        float gg = tanhf(pre[u * 4 + 3]);
        int j = j0 + u;
        float cn = fg * cbuf[j] + ig * gg;
        float hn = og * tanhf(cn);
        cbuf[j] = cn;
        __bf16 hb = (__bf16)hn;
        hbuf[((t + 1) & 1) * HIDW + j] = hb;      // next step's input
        hs[(size_t)t * HIDW + j] = hb;            // for output GEMM
        if (t == TLEN - 1) { hc_fin[j] = hn; hc_fin[HIDW + j] = cn; }
      }
    }
    grid_sync(bar, NBLK);  // read-before-write for double buffer + visibility
  }
}

// --------------------- out = hs @ WoutT^T + bout (WMMA) --------------------
__global__ void k_outgemm(const __bf16* __restrict__ hs, const __bf16* __restrict__ WoutT,
                          const float* __restrict__ bout, float* __restrict__ out) {
  int wave = threadIdx.x >> 5, lane = threadIdx.x & 31;
  int half = lane >> 4, l15 = lane & 15;
  int nbase = (blockIdx.x * 8 + wave) * 16;   // [0, 256)
  int tbase = blockIdx.y * 16;                // [0, 4096)

  v8f acc = {};
  const __bf16* arow = hs    + (size_t)(tbase + l15) * HIDW + half * 8;
  const __bf16* brow = WoutT + (size_t)(nbase + l15) * HIDW + half * 8;
#pragma unroll 8
  for (int kc = 0; kc < HIDW / 32; ++kc) {
    Frag a, b;
    a.h[0] = *(const v8bf*)(arow);
    a.h[1] = *(const v8bf*)(arow + 16);
    b.h[0] = *(const v8bf*)(brow);
    b.h[1] = *(const v8bf*)(brow + 16);
    acc = wmma_bf16(acc, a, b);
    arow += 32; brow += 32;
  }
  int n = nbase + l15;
  float bias = bout[n];
#pragma unroll
  for (int q = 0; q < 8; ++q) {
    int row = tbase + q + half * 8;
    out[(size_t)row * NCH + n] = acc[q] + bias;
  }
}

// ---------------------------------------------------------------------------
extern "C" void kernel_launch(void* const* d_in, const int* in_sizes, int n_in,
                              void* d_out, int out_size, void* d_ws, size_t ws_size,
                              hipStream_t stream) {
  const int*   seq    = (const int*)  d_in[0];
  const float* hidden = (const float*)d_in[1];
  const float* cell   = (const float*)d_in[2];
  const float* emb    = (const float*)d_in[3];
  const float* Wf     = (const float*)d_in[4];
  const float* bfv    = (const float*)d_in[5];
  const float* Wi     = (const float*)d_in[6];
  const float* biv    = (const float*)d_in[7];
  const float* Wo     = (const float*)d_in[8];
  const float* bov    = (const float*)d_in[9];
  const float* Wc     = (const float*)d_in[10];
  const float* bcv    = (const float*)d_in[11];
  const float* Wout   = (const float*)d_in[12];
  const float* bout   = (const float*)d_in[13];
  float* out = (float*)d_out;  // [T*256] out, then h_fin[2048], c_fin[2048]

  // workspace carve-up (~131 MB, 256-B aligned chunks)
  char* p = (char*)d_ws;
  unsigned* bar   = (unsigned*)p;            p += 256;
  __bf16*   hbuf  = (__bf16*)p;              p += (size_t)2 * HIDW * 2;      // 8 KB
  float*    cbuf  = (float*)p;               p += (size_t)HIDW * 4;          // 8 KB
  __bf16*   WxT   = (__bf16*)p;              p += (size_t)GATE * EMBW * 2;   // 8 MB
  __bf16*   WhT   = (__bf16*)p;              p += (size_t)GATE * HIDW * 2;   // 32 MB
  __bf16*   WoutT = (__bf16*)p;              p += (size_t)NCH * HIDW * 2;    // 1 MB
  __bf16*   X     = (__bf16*)p;              p += (size_t)TLEN * EMBW * 2;   // 4 MB
  __bf16*   Xp    = (__bf16*)p;              p += (size_t)TLEN * GATE * 2;   // 64 MB
  __bf16*   hs    = (__bf16*)p;              p += (size_t)TLEN * HIDW * 2;   // 16 MB

  k_init<<<8, 256, 0, stream>>>(hidden, cell, hbuf, cbuf, bar);
  k_convert_gates<<<(GATE * (size_t)DIN) / 256, 256, 0, stream>>>(Wf, Wi, Wo, Wc, WxT, WhT);
  k_convert_wout<<<((size_t)NCH * HIDW) / 256, 256, 0, stream>>>(Wout, WoutT);
  k_embed<<<((size_t)TLEN * EMBW) / 256, 256, 0, stream>>>(seq, emb, X);
  k_xproj<<<dim3(GATE / 128, TLEN / 16), 256, 0, stream>>>(X, WxT, bfv, biv, bov, bcv, Xp);
  k_recurrent<<<NBLK, 256, 0, stream>>>(WhT, Xp, hbuf, cbuf, hs,
                                        out + (size_t)TLEN * NCH, bar);
  k_outgemm<<<dim3(NCH / 128, TLEN / 16), 256, 0, stream>>>(hs, WoutT, bout, out);
}
